// MoELayer_20830591386389
// MI455X (gfx1250) — compile-verified, matrix operand order checked by reading, not run on hardware
//
#include <hip/hip_runtime.h>

// ---------------- problem constants ----------------
#define B_    4
#define S_    1600
#define D_    768
#define E_    16
#define H_    3072
#define T_    (B_ * S_)      // 6400 tokens

// ---------------- WMMA types ----------------
typedef __attribute__((ext_vector_type(16))) __bf16 v16bf;
typedef __attribute__((ext_vector_type(8)))  float  v8f;

union ABFrag { unsigned int u[8]; v16bf v; };

__device__ __forceinline__ unsigned short f2bf(float f) {
  union { float f; unsigned int u; } c; c.f = f;
  unsigned int u = c.u;
  u += 0x7FFFu + ((u >> 16) & 1u);           // round-to-nearest-even
  return (unsigned short)(u >> 16);
}

// load one 16x32 bf16 fragment (ISA 16-bit K-pair packing) from a pointer
// whose K dimension is contiguous. kg = lane>>4 selects the K half-group.
__device__ __forceinline__ void load_frag(ABFrag& f, const unsigned short* p, int k0, int kg) {
#pragma unroll
  for (int j = 0; j < 8; ++j) {
    int kk = k0 + 2 * (j & 3) + 16 * (j >> 2) + 8 * kg;
    f.u[j] = *(const unsigned int*)&p[kk];
  }
}

// ---------------- tiny init / zero kernels ----------------
__global__ void moe_init(int* cnt, float* sum_prob, float* sum_topk) {
  int i = threadIdx.x;
  if (i < E_) { cnt[i] = 0; sum_prob[i] = 0.f; sum_topk[i] = 0.f; }
}

__global__ void moe_zero(float* p, int n) {
  for (int i = blockIdx.x * blockDim.x + threadIdx.x; i < n; i += gridDim.x * blockDim.x)
    p[i] = 0.f;
}

// ---------------- one-time weight conversion (fp32 -> bf16, transposed) ----
// w1  : [E][D][H]  ->  w1t : [E][H][D]   (k = d contiguous per output column)
__global__ void moe_conv_w1t(const float* __restrict__ w1, unsigned short* __restrict__ w1t) {
  const int total = E_ * D_ * H_;
  for (int i = blockIdx.x * blockDim.x + threadIdx.x; i < total; i += gridDim.x * blockDim.x) {
    int d = i % D_; int r = i / D_; int h = r % H_; int e = r / H_;
    w1t[i] = f2bf(w1[((long long)e * D_ + d) * H_ + h]);
  }
}
// w2  : [E][H][D]  ->  w2t : [E][D][H]   (k = h contiguous per output column)
__global__ void moe_conv_w2t(const float* __restrict__ w2, unsigned short* __restrict__ w2t) {
  const int total = E_ * D_ * H_;
  for (int i = blockIdx.x * blockDim.x + threadIdx.x; i < total; i += gridDim.x * blockDim.x) {
    int h = i % H_; int r = i / H_; int d = r % D_; int e = r / D_;
    w2t[i] = f2bf(w2[((long long)e * H_ + h) * D_ + d]);
  }
}

// ---------------- gating: one wave32 per token ----------------
__global__ __launch_bounds__(128) void moe_gate(
    const float* __restrict__ x, const float* __restrict__ gw,
    int* __restrict__ cnt, int* __restrict__ list, float* __restrict__ wlist,
    float* __restrict__ sum_prob, float* __restrict__ sum_topk)
{
  const int lane = threadIdx.x & 31;
  const int wave = threadIdx.x >> 5;
  const int t = blockIdx.x * 4 + wave;
  if (t >= T_) return;

  float acc[E_];
#pragma unroll
  for (int e = 0; e < E_; ++e) acc[e] = 0.f;

  const float* xr = x + (long long)t * D_;
  for (int d = lane; d < D_; d += 32) {
    float xv = xr[d];
#pragma unroll
    for (int e = 0; e < E_; ++e) acc[e] += xv * gw[e * D_ + d];
  }
  // wave-wide all-reduce of the 16 partial dots
#pragma unroll
  for (int e = 0; e < E_; ++e) {
#pragma unroll
    for (int off = 16; off > 0; off >>= 1)
      acc[e] += __shfl_xor(acc[e], off, 32);
  }
  // softmax over 16 logits (every lane redundantly)
  float mx = acc[0];
#pragma unroll
  for (int e = 1; e < E_; ++e) mx = fmaxf(mx, acc[e]);
  float p[E_]; float sum = 0.f;
#pragma unroll
  for (int e = 0; e < E_; ++e) { p[e] = __expf(acc[e] - mx); sum += p[e]; }
  float inv = 1.f / sum;
#pragma unroll
  for (int e = 0; e < E_; ++e) p[e] *= inv;

  // top-2 (lowest index wins ties, like lax.top_k)
  int i1 = 0; float v1 = p[0];
#pragma unroll
  for (int e = 1; e < E_; ++e) if (p[e] > v1) { v1 = p[e]; i1 = e; }
  int i2 = (i1 == 0) ? 1 : 0; float v2 = p[i2];
#pragma unroll
  for (int e = 0; e < E_; ++e) if (e != i1 && p[e] > v2) { v2 = p[e]; i2 = e; }

  float wn = 1.f / (v1 + v2);
  if (lane == 0) {
    int pos1 = atomicAdd(&cnt[i1], 1);
    list[i1 * T_ + pos1] = t; wlist[i1 * T_ + pos1] = v1 * wn;
    int pos2 = atomicAdd(&cnt[i2], 1);
    list[i2 * T_ + pos2] = t; wlist[i2 * T_ + pos2] = v2 * wn;
  }
  if (lane < E_) {
    atomicAdd(&sum_prob[lane], p[lane]);
    atomicAdd(&sum_topk[lane], (p[lane] > v2) ? 1.f : 0.f);
  }
}

// ---------------- fused expert FFN: relu(X@W1+b1)@W2+b2, weighted scatter ---
// 8 waves / block, M-tile = 32 tokens. Each wave owns a 2x4 WMMA micro-tile
// (both 16-row sub-tiles x 4 N-tiles). K-loop unrolled by 64 with ping-pong
// double-buffered B fragments so global weight loads overlap 8 in-flight
// WMMAs. Weight stream read once per block.
#define MTILE 32
#define XSTR  776     // padded bf16 row stride for X tile in LDS
#define HSTR  3080    // padded bf16 row stride for H tile in LDS

__global__ __launch_bounds__(256) void moe_ffn(
    const float* __restrict__ x,
    const unsigned short* __restrict__ w1t,  // [E][H][D] bf16
    const float* __restrict__ b1,            // [E][H]
    const unsigned short* __restrict__ w2t,  // [E][D][H] bf16
    const float* __restrict__ b2,            // [E][D]
    const int* __restrict__ cnt,
    const int* __restrict__ list, const float* __restrict__ wlist,
    float* __restrict__ out)
{
  extern __shared__ char smem[];
  int*            tok = (int*)smem;                                       // 32 ints
  float*          wgt = (float*)(smem + 128);                             // 32 floats
  unsigned short* Xs  = (unsigned short*)(smem + 256);                    // 32 x XSTR bf16
  unsigned short* Hs  = (unsigned short*)(smem + 256 + MTILE * XSTR * 2); // 32 x HSTR bf16

  const int e    = blockIdx.y;
  const int tile = blockIdx.x;
  const int nTok = cnt[e];
  const int base = tile * MTILE;
  if (base >= nTok) return;

  const int tid  = threadIdx.x;
  const int lane = tid & 31;
  const int wave = tid >> 5;

  if (tid < MTILE) {
    int m = tid;
    if (base + m < nTok) { tok[m] = list[e * T_ + base + m]; wgt[m] = wlist[e * T_ + base + m]; }
    else                 { tok[m] = -1;                       wgt[m] = 0.f; }
  }
  __syncthreads();

  // stage X tile (gather rows, fp32 -> bf16)
  for (int i = tid; i < MTILE * D_; i += 256) {
    int m = i / D_, d = i - m * D_;
    int tk = tok[m];
    float v = (tk >= 0) ? x[(long long)tk * D_ + d] : 0.f;
    Xs[m * XSTR + d] = f2bf(v);
  }
  __syncthreads();

  const int m0 = lane & 15;  // lane's M (A) / N (B,C) index
  const int kg = lane >> 4;  // K half-group per ISA 16-bit layout
  const unsigned short* xr0 = Xs + m0 * XSTR;         // A row, sub-tile r=0
  const unsigned short* xr1 = Xs + (16 + m0) * XSTR;  // A row, sub-tile r=1
  const unsigned short* hr0 = Hs + m0 * HSTR;
  const unsigned short* hr1 = Hs + (16 + m0) * HSTR;

  // -------- phase 1: Hs = relu(X @ W1 + b1) --------
  const unsigned short* w1e = w1t + (long long)e * H_ * D_;
  for (int q = wave; q < H_ / 64; q += 8) {        // 48 column-quads, 6 per wave
    const int nb = q * 64 + m0;
    const unsigned short* bc[4];
#pragma unroll
    for (int s = 0; s < 4; ++s) bc[s] = w1e + (long long)(nb + 16 * s) * D_;
    v8f acc[8];                                     // acc[2*s + r]
#pragma unroll
    for (int i = 0; i < 8; ++i) acc[i] = (v8f){};
    ABFrag b0[4], b1f[4];
#pragma unroll
    for (int s = 0; s < 4; ++s) load_frag(b0[s], bc[s], 0, kg);
    for (int k0 = 0; k0 < D_; k0 += 64) {           // 12 iters, 2 K-steps each
      // -- K-step A: consume b0, load b1f for k0+32 --
#pragma unroll
      for (int s = 0; s < 4; ++s) {
        __builtin_prefetch(bc[s] + k0 + 256, 0, 1); // 4 lines ahead, 1 line/iter
        load_frag(b1f[s], bc[s], k0 + 32, kg);
      }
      {
        ABFrag a0, a1;
        load_frag(a0, xr0, k0, kg);
        load_frag(a1, xr1, k0, kg);
#pragma unroll
        for (int s = 0; s < 4; ++s) {
          acc[2*s]   = __builtin_amdgcn_wmma_f32_16x16x32_bf16(false, a0.v, false, b0[s].v,
                                                               (short)0, acc[2*s],   false, false);
          acc[2*s+1] = __builtin_amdgcn_wmma_f32_16x16x32_bf16(false, a1.v, false, b0[s].v,
                                                               (short)0, acc[2*s+1], false, false);
        }
      }
      // -- K-step B: consume b1f, load b0 for k0+64 --
      if (k0 + 64 < D_) {
#pragma unroll
        for (int s = 0; s < 4; ++s) load_frag(b0[s], bc[s], k0 + 64, kg);
      }
      {
        ABFrag a0, a1;
        load_frag(a0, xr0, k0 + 32, kg);
        load_frag(a1, xr1, k0 + 32, kg);
#pragma unroll
        for (int s = 0; s < 4; ++s) {
          acc[2*s]   = __builtin_amdgcn_wmma_f32_16x16x32_bf16(false, a0.v, false, b1f[s].v,
                                                               (short)0, acc[2*s],   false, false);
          acc[2*s+1] = __builtin_amdgcn_wmma_f32_16x16x32_bf16(false, a1.v, false, b1f[s].v,
                                                               (short)0, acc[2*s+1], false, false);
        }
      }
    }
#pragma unroll
    for (int s = 0; s < 4; ++s) {
      const int n = nb + 16 * s;
      const float bb = b1[e * H_ + n];
#pragma unroll
      for (int j = 0; j < 8; ++j) {
        int m = j + 8 * kg;                         // C layout: VGPR j -> row j (+8 upper half)
        float v0 = acc[2*s][j]   + bb; v0 = v0 > 0.f ? v0 : 0.f;
        float v1 = acc[2*s+1][j] + bb; v1 = v1 > 0.f ? v1 : 0.f;
        Hs[m * HSTR + n]        = f2bf(v0);
        Hs[(16 + m) * HSTR + n] = f2bf(v1);
      }
    }
  }
  __syncthreads();

  // -------- phase 2: out += diag(w) * (Hs @ W2 + b2) --------
  const unsigned short* w2e = w2t + (long long)e * D_ * H_;
  for (int q = wave; q < D_ / 64; q += 8) {        // 12 column-quads
    const int nb = q * 64 + m0;
    const unsigned short* bc[4];
#pragma unroll
    for (int s = 0; s < 4; ++s) bc[s] = w2e + (long long)(nb + 16 * s) * H_;
    v8f acc[8];
#pragma unroll
    for (int i = 0; i < 8; ++i) acc[i] = (v8f){};
    ABFrag b0[4], b1f[4];
#pragma unroll
    for (int s = 0; s < 4; ++s) load_frag(b0[s], bc[s], 0, kg);
    for (int k0 = 0; k0 < H_; k0 += 64) {           // 48 iters, 2 K-steps each
#pragma unroll
      for (int s = 0; s < 4; ++s) {
        __builtin_prefetch(bc[s] + k0 + 256, 0, 1);
        load_frag(b1f[s], bc[s], k0 + 32, kg);
      }
      {
        ABFrag a0, a1;
        load_frag(a0, hr0, k0, kg);
        load_frag(a1, hr1, k0, kg);
#pragma unroll
        for (int s = 0; s < 4; ++s) {
          acc[2*s]   = __builtin_amdgcn_wmma_f32_16x16x32_bf16(false, a0.v, false, b0[s].v,
                                                               (short)0, acc[2*s],   false, false);
          acc[2*s+1] = __builtin_amdgcn_wmma_f32_16x16x32_bf16(false, a1.v, false, b0[s].v,
                                                               (short)0, acc[2*s+1], false, false);
        }
      }
      if (k0 + 64 < H_) {
#pragma unroll
        for (int s = 0; s < 4; ++s) load_frag(b0[s], bc[s], k0 + 64, kg);
      }
      {
        ABFrag a0, a1;
        load_frag(a0, hr0, k0 + 32, kg);
        load_frag(a1, hr1, k0 + 32, kg);
#pragma unroll
        for (int s = 0; s < 4; ++s) {
          acc[2*s]   = __builtin_amdgcn_wmma_f32_16x16x32_bf16(false, a0.v, false, b1f[s].v,
                                                               (short)0, acc[2*s],   false, false);
          acc[2*s+1] = __builtin_amdgcn_wmma_f32_16x16x32_bf16(false, a1.v, false, b1f[s].v,
                                                               (short)0, acc[2*s+1], false, false);
        }
      }
    }
#pragma unroll
    for (int s = 0; s < 4; ++s) {
      const int n = nb + 16 * s;
      const float bb = b2[e * D_ + n];
#pragma unroll
      for (int j = 0; j < 8; ++j) {
        int m = j + 8 * kg;
        int tk0 = tok[m];
        int tk1 = tok[16 + m];
        if (tk0 >= 0) atomicAdd(&out[(long long)tk0 * D_ + n], wgt[m]      * (acc[2*s][j]   + bb));
        if (tk1 >= 0) atomicAdd(&out[(long long)tk1 * D_ + n], wgt[16 + m] * (acc[2*s+1][j] + bb));
      }
    }
  }
}

// ---------------- aux loss finalize ----------------
__global__ void moe_aux(const float* sum_prob, const float* sum_topk, float* aux_out) {
  if (threadIdx.x == 0 && blockIdx.x == 0) {
    const float inv = 1.f / (float)T_;
    float l = 0.f;
    for (int e = 0; e < E_; ++e) l += (sum_prob[e] * inv) * (sum_topk[e] * inv);
    *aux_out = l * (float)E_;
  }
}

// ---------------- launcher ----------------
extern "C" void kernel_launch(void* const* d_in, const int* in_sizes, int n_in,
                              void* d_out, int out_size, void* d_ws, size_t ws_size,
                              hipStream_t stream)
{
  (void)in_sizes; (void)n_in; (void)out_size; (void)ws_size;
  const float* x  = (const float*)d_in[0];
  const float* gw = (const float*)d_in[1];
  const float* w1 = (const float*)d_in[2];
  const float* b1 = (const float*)d_in[3];
  const float* w2 = (const float*)d_in[4];
  const float* b2 = (const float*)d_in[5];
  float* out = (float*)d_out;

  // workspace layout
  char* ws = (char*)d_ws;
  int*   cnt      = (int*)(ws + 0);                                   // E ints
  float* sum_prob = (float*)(ws + 64);                                // E floats
  float* sum_topk = (float*)(ws + 128);                               // E floats
  int*   list     = (int*)(ws + 256);                                 // E*T ints
  float* wlist    = (float*)(ws + 256 + (size_t)E_ * T_ * 4);         // E*T floats
  unsigned short* w1t = (unsigned short*)(ws + (1u << 20));           // bf16 [E][H][D]
  unsigned short* w2t = w1t + (size_t)E_ * H_ * D_;                   // bf16 [E][D][H]
  // total ws need ~= 1MB + 2*75.5MB = ~152MB

  moe_init<<<1, 64, 0, stream>>>(cnt, sum_prob, sum_topk);
  moe_zero<<<2048, 256, 0, stream>>>(out, T_ * D_);
  moe_conv_w1t<<<4096, 256, 0, stream>>>(w1, w1t);
  moe_conv_w2t<<<4096, 256, 0, stream>>>(w2, w2t);
  moe_gate<<<T_ / 4, 128, 0, stream>>>(x, gw, cnt, list, wlist, sum_prob, sum_topk);

  const size_t smem = 256 + (size_t)MTILE * XSTR * 2 + (size_t)MTILE * HSTR * 2; // ~247 KB (fits 320 KB WGP LDS)
  dim3 grid((T_ + MTILE - 1) / MTILE, E_);  // worst case: one expert gets all tokens
  moe_ffn<<<grid, 256, smem, stream>>>(x, w1t, b1, w2t, b2, cnt, list, wlist, out);

  moe_aux<<<1, 32, 0, stream>>>(sum_prob, sum_topk, out + (size_t)T_ * D_);
}
